// ParameterPredictorGNNTimeBeta_24043226923277
// MI455X (gfx1250) — matrix operand order repaired and checked by reading, not compiled, not run on hardware
//
#include <hip/hip_runtime.h>
#include <hip/hip_bf16.h>

typedef unsigned short ushort8 __attribute__((ext_vector_type(8)));
typedef __bf16 v16bf __attribute__((ext_vector_type(16)));
typedef float v8f __attribute__((ext_vector_type(8)));

union Frag {
    ushort8 h[2];
    v16bf v;
};

__device__ __forceinline__ unsigned short f2bf(float f) {
    unsigned int u = __float_as_uint(f);
    unsigned int r = u + 0x7FFFu + ((u >> 16) & 1u);   // round-to-nearest-even
    return (unsigned short)(r >> 16);
}

// ---------------------------------------------------------------- degree prep
__global__ __launch_bounds__(256) void k_init_deg(float* deg, int n) {
    int i = blockIdx.x * 256 + threadIdx.x;
    if (i < n) deg[i] = 1.0f;                 // self-loop weight
}

__global__ __launch_bounds__(256) void k_deg_accum(const long long* __restrict__ col,
                                                   const float* __restrict__ ew,
                                                   float* __restrict__ deg, int E) {
    int e = blockIdx.x * 256 + threadIdx.x;
    if (e < E) atomicAdd(&deg[(int)col[e]], ew[e]);
}

__global__ __launch_bounds__(256) void k_dinv(float* deg, int n) {
    int i = blockIdx.x * 256 + threadIdx.x;
    if (i < n) {
        float d = deg[i];
        deg[i] = (d > 0.0f) ? rsqrtf(d) : 0.0f;
    }
}

// ---------------------------------------------------------------- WMMA GEMM
// C[M, NC] = (relu?)(A[M,128]) @ W[128, NC], bf16 inputs, f32 accumulate.
// Block: 256 threads = 8 wave32, 64 rows per block. Each wave: one 16-row
// tile, NTW N-tiles processed as independent pairs (two acc chains) to fill
// the WMMA->WMMA co-exec hazard slots.
template <int NC, bool RELU>
__global__ __launch_bounds__(256) void k_gemm_wmma(const float* __restrict__ A,
                                                   const float* __restrict__ W,
                                                   float* __restrict__ C, int M) {
    static_assert(NC % 32 == 0, "NC must give an even number of 16-wide N tiles per wave");
    __shared__ unsigned short Alds[64][128];   // [row][k]     16 KB
    __shared__ unsigned short Wlds[NC][128];   // [n][k] (W^T) <=32 KB

    const int tid = threadIdx.x;
    const int rowBase = blockIdx.x * 64;
    const bool fullTile = (rowBase + 64 <= M);

    // Stage A tile (f32 -> bf16), optional fused ReLU (layer-2 input).
    if (fullTile) {
        for (int i = tid; i < 64 * 128; i += 256) {
            int r = i >> 7, c = i & 127;
            float v = A[(size_t)(rowBase + r) * 128 + c];
            if (RELU) v = fmaxf(v, 0.0f);
            Alds[r][c] = f2bf(v);
        }
    } else {
        for (int i = tid; i < 64 * 128; i += 256) {
            int r = i >> 7, c = i & 127;
            int gr = rowBase + r;
            float v = (gr < M) ? A[(size_t)gr * 128 + c] : 0.0f;
            if (RELU) v = fmaxf(v, 0.0f);
            Alds[r][c] = f2bf(v);
        }
    }
    // Stage W transposed so B fragments are contiguous in K.
    for (int i = tid; i < NC * 128; i += 256) {
        int k = i / NC, n = i % NC;
        Wlds[n][k] = f2bf(W[(size_t)k * NC + n]);
    }
    __syncthreads();

    const int lane = tid & 31;
    const int wave = tid >> 5;
    const int rt = wave & 3;        // row tile 0..3
    const int nhalf = wave >> 2;    // N-range half
    const int NTW = (NC / 16) / 2;  // N tiles per wave (4 or 2)

    // A fragments for this wave's 16-row tile, one per K-step (K=128 -> 4).
    // ISA 16-bit A 16x32 layout: lanes 0-15 hold K 0..7 & 16..23, lanes 16-31
    // hold K 8..15 & 24..31 (row M = lane%16).
    Frag a[4];
    {
        int row = rt * 16 + (lane & 15);
        int kb = (lane < 16) ? 0 : 8;
#pragma unroll
        for (int kt = 0; kt < 4; ++kt) {
            int k0 = kt * 32 + kb;
            a[kt].h[0] = *reinterpret_cast<const ushort8*>(&Alds[row][k0]);
            a[kt].h[1] = *reinterpret_cast<const ushort8*>(&Alds[row][k0 + 16]);
        }
    }

    const int kbB = (lane < 16) ? 0 : 8;   // B 32x16: lanes 0-15 K 0..7 & 16..23, 16-31 K 8..15 & 24..31
    for (int j = 0; j < NTW; j += 2) {
        int nt0 = nhalf * NTW + j;          // two adjacent 16-col tiles
        int n0 = nt0 * 16 + (lane & 15);
        int n1 = n0 + 16;
        v8f acc0 = {};
        v8f acc1 = {};
#pragma unroll
        for (int kt = 0; kt < 4; ++kt) {
            int k0 = kt * 32 + kbB;
            Frag b0, b1;
            b0.h[0] = *reinterpret_cast<const ushort8*>(&Wlds[n0][k0]);
            b0.h[1] = *reinterpret_cast<const ushort8*>(&Wlds[n0][k0 + 16]);
            b1.h[0] = *reinterpret_cast<const ushort8*>(&Wlds[n1][k0]);
            b1.h[1] = *reinterpret_cast<const ushort8*>(&Wlds[n1][k0 + 16]);
            // Two independent accumulator chains hide the WMMA RAW hazard.
            acc0 = __builtin_amdgcn_wmma_f32_16x16x32_bf16(
                false, a[kt].v, false, b0.v, (short)0, acc0, false, false);
            acc1 = __builtin_amdgcn_wmma_f32_16x16x32_bf16(
                false, a[kt].v, false, b1.v, (short)0, acc1, false, false);
        }
        // C/D layout: VGPR r -> M = r + (lane/16)*8, N = nt*16 + lane%16.
        int mBase = rowBase + rt * 16 + ((lane >> 4) << 3);
        float* p = C + (size_t)mBase * NC + nt0 * 16 + (lane & 15);
        if (fullTile) {
#pragma unroll
            for (int r = 0; r < 8; ++r) {       // constant byte offsets, no guards
                p[(size_t)r * NC] = acc0[r];
                p[(size_t)r * NC + 16] = acc1[r];
            }
        } else {
#pragma unroll
            for (int r = 0; r < 8; ++r) {
                if (mBase + r < M) {
                    p[(size_t)r * NC] = acc0[r];
                    p[(size_t)r * NC + 16] = acc1[r];
                }
            }
        }
    }
}

// ----------------------------------------------- aggregation: bias + self-loop
__global__ __launch_bounds__(256) void k_agg_init(const float* __restrict__ xw,
                                                  const float* __restrict__ dinv,
                                                  const float* __restrict__ bias,
                                                  float* __restrict__ agg,
                                                  int n, int F) {
    long long gid = (long long)blockIdx.x * 256 + threadIdx.x;
    int cpn = F >> 2;
    long long node = gid / cpn;
    int cg = (int)(gid % cpn);
    if (node >= n) return;
    float s = dinv[node];
    float s2 = s * s;   // self-loop norm: dinv * 1 * dinv
    const float4 v = *reinterpret_cast<const float4*>(xw + (size_t)node * F + cg * 4);
    const float4 b = *reinterpret_cast<const float4*>(bias + cg * 4);
    float4 o;
    o.x = b.x + s2 * v.x;
    o.y = b.y + s2 * v.y;
    o.z = b.z + s2 * v.z;
    o.w = b.w + s2 * v.w;
    *reinterpret_cast<float4*>(agg + (size_t)node * F + cg * 4) = o;
}

// ------------------------------------------------ edge-parallel scatter (L2 atomics)
__global__ __launch_bounds__(256) void k_scatter(const long long* __restrict__ rowIdx,
                                                 const long long* __restrict__ colIdx,
                                                 const float* __restrict__ ew,
                                                 const float* __restrict__ dinv,
                                                 const float* __restrict__ xw,
                                                 float* __restrict__ agg,
                                                 int E, int F) {
    long long gid = (long long)blockIdx.x * 256 + threadIdx.x;
    int lpe = F >> 2;                 // lanes per edge (float4 per lane)
    long long e = gid / lpe;
    int cg = (int)(gid % lpe);
    if (e >= E) return;
    int r = (int)rowIdx[e];
    int c = (int)colIdx[e];
    float nrm = dinv[r] * ew[e] * dinv[c];
    const float4 v = *reinterpret_cast<const float4*>(xw + (size_t)r * F + cg * 4);
    float* dst = agg + (size_t)c * F + cg * 4;
    atomicAdd(dst + 0, nrm * v.x);
    atomicAdd(dst + 1, nrm * v.y);
    atomicAdd(dst + 2, nrm * v.z);
    atomicAdd(dst + 3, nrm * v.w);
}

// ------------------------------------------------------------- output heads
__global__ __launch_bounds__(256) void k_heads(const float* __restrict__ h,
                                               const float* __restrict__ Wbi, const float* __restrict__ bbi,
                                               const float* __restrict__ Wbd, const float* __restrict__ bbd,
                                               const float* __restrict__ Wg,  const float* __restrict__ bg,
                                               float* __restrict__ out, int N) {
    int wave = threadIdx.x >> 5;
    int lane = threadIdx.x & 31;
    int node = blockIdx.x * 8 + wave;
    if (node >= N) return;   // uniform per wave
    const float* hp = h + (size_t)node * 64 + lane * 2;
    float h0 = fmaxf(hp[0], 0.0f);
    float h1 = fmaxf(hp[1], 0.0f);
    int f = lane * 2;
    float pi = h0 * Wbi[f] + h1 * Wbi[f + 1];
    float pd = h0 * Wbd[f] + h1 * Wbd[f + 1];
    float pg = h0 * Wg[f] + h1 * Wg[f + 1];
#pragma unroll
    for (int m = 16; m > 0; m >>= 1) {
        pi += __shfl_xor(pi, m, 32);
        pd += __shfl_xor(pd, m, 32);
        pg += __shfl_xor(pg, m, 32);
    }
    if (lane == 0) {
        float bi = 2.0f / (1.0f + __expf(-(pi + bbi[0])));
        float bd = 0.1f / (1.0f + __expf(-(pd + bbd[0])));
        float g  = 0.3f / (1.0f + __expf(-(pg + bg[0])));
        out[node]         = fminf(fmaxf(bi, 1e-4f), 2.0f);
        out[N + node]     = fminf(fmaxf(bd, 1e-6f), 0.1f);
        out[2 * N + node] = fminf(fmaxf(g,  1e-4f), 0.3f);
    }
}

// ---------------------------------------------------------------- launcher
extern "C" void kernel_launch(void* const* d_in, const int* in_sizes, int n_in,
                              void* d_out, int out_size, void* d_ws, size_t ws_size,
                              hipStream_t stream) {
    const float*      x   = (const float*)d_in[0];
    const long long*  ei  = (const long long*)d_in[1];   // int64 [2, E]
    const float*      ew  = (const float*)d_in[2];
    const float*      W1  = (const float*)d_in[3];
    const float*      b1  = (const float*)d_in[4];
    const float*      W2  = (const float*)d_in[5];
    const float*      b2  = (const float*)d_in[6];
    const float*      Wbi = (const float*)d_in[7];
    const float*      bbi = (const float*)d_in[8];
    const float*      Wbd = (const float*)d_in[9];
    const float*      bbd = (const float*)d_in[10];
    const float*      Wg  = (const float*)d_in[11];
    const float*      bg  = (const float*)d_in[12];
    float* out = (float*)d_out;

    const int N = in_sizes[0] / 128;
    const int E = in_sizes[2];
    const long long* rowI = ei;       // edge_index[0]
    const long long* colI = ei + E;   // edge_index[1]

    float* ws   = (float*)d_ws;
    float* dinv = ws;                                   // N floats (deg -> dinv)
    float* bufA = ws + N;                               // N*128  (xw1, then xw2)
    float* bufB = bufA + (size_t)N * 128;               // N*128  (agg1, then agg2)

    const int TB = 256;
    auto cdiv = [](long long a, long long b) { return (unsigned)((a + b - 1) / b); };

    // degree / normalization
    k_init_deg<<<cdiv(N, TB), TB, 0, stream>>>(dinv, N);
    k_deg_accum<<<cdiv(E, TB), TB, 0, stream>>>(colI, ew, dinv, E);
    k_dinv<<<cdiv(N, TB), TB, 0, stream>>>(dinv, N);

    // layer 1: xw1 = x @ W1 ; agg1 = b1 + dinv^2*xw1 + scatter ; relu fused into next GEMM
    k_gemm_wmma<128, false><<<cdiv(N, 64), TB, 0, stream>>>(x, W1, bufA, N);
    k_agg_init<<<cdiv((long long)N * 32, TB), TB, 0, stream>>>(bufA, dinv, b1, bufB, N, 128);
    k_scatter<<<cdiv((long long)E * 32, TB), TB, 0, stream>>>(rowI, colI, ew, dinv, bufA, bufB, E, 128);

    // layer 2: xw2 = relu(agg1) @ W2 ; agg2 = b2 + dinv^2*xw2 + scatter
    k_gemm_wmma<64, true><<<cdiv(N, 64), TB, 0, stream>>>(bufB, W2, bufA, N);
    k_agg_init<<<cdiv((long long)N * 16, TB), TB, 0, stream>>>(bufA, dinv, b2, bufB, N, 64);
    k_scatter<<<cdiv((long long)E * 16, TB), TB, 0, stream>>>(rowI, colI, ew, dinv, bufA, bufB, E, 64);

    // heads: relu + 3x (64->1) dot, sigmoid, scale, clip
    k_heads<<<cdiv(N, 8), TB, 0, stream>>>(bufB, Wbi, bbi, Wbd, bbd, Wg, bg, out, N);
}
